// Model_17274358465006
// MI455X (gfx1250) — matrix operand order repaired and checked by reading, not compiled, not run on hardware
//
#include <hip/hip_runtime.h>
#include <stdint.h>

typedef __attribute__((ext_vector_type(16))) _Float16 v16h;
typedef __attribute__((ext_vector_type(8)))  float    v8f;

#define BB    16
#define NVARS 7
#define PP    96
#define NTOK  97
#define LL    679
#define DDIM  512
#define HH    8
#define EE    64
#define DFF   2048
#define MSROW (BB*LL)   // 10864

// ---------------- WMMA GEMM: C[M,N] = A[M,K] @ Wt[N,K]^T + bias, opt ReLU ----
// A and Wt are f16 in memory (Wt pre-transposed), so staging is a pure copy.
// 128x128 block tile, BK=32, 8 waves each 32(M) x 64(N) = 2x4 WMMA frags.
// Staging pipeline uses named scalar uint4s (NOT arrays) so prefetch data stays
// in VGPRs -- avoids AMDGPUPromoteAlloca demoting them to LDS.
#define BM 128
#define BN 128
#define BK 32
#define LDT 40   // LDS row stride in halves: 80B rows (16B-aligned b128 stores),
                 // 20-dword lane stride -> conflict-free; frag gathers = 2x b128

__global__ __launch_bounds__(256)
void gemm_bias_kernel(const _Float16* __restrict__ A, const _Float16* __restrict__ Wt,
                      const float* __restrict__ bias,
                      float* __restrict__ Cf, _Float16* __restrict__ Ch,
                      int M, int N, int K, int relu)
{
    __shared__ _Float16 As[BM][LDT];   // A tile (m, k)
    __shared__ _Float16 Bt[BN][LDT];   // B tile (n, k)

    const int tid    = threadIdx.x;
    const int lane   = tid & 31;
    const int wave   = tid >> 5;
    const int wm     = wave >> 1;       // 0..3  (M)
    const int wn     = wave & 1;        // 0..1  (N)
    const int laneLo = lane & 15;
    const int laneHi = lane >> 4;

    const int nBase = blockIdx.x * BN;
    const int mBase = blockIdx.y * BM;

    // per-thread staging coordinates (two 8-half chunks)
    const int e0 = tid * 8,         row0 = e0 >> 5, kk0 = e0 & 31;
    const int e1 = (256 + tid) * 8, row1 = e1 >> 5, kk1 = e1 & 31;
    int gr0 = mBase + row0; if (gr0 > M-1) gr0 = M-1;     // clamp, no divergence
    int gr1 = mBase + row1; if (gr1 > M-1) gr1 = M-1;
    const _Float16* pa0 = A  + (size_t)gr0*K          + kk0;
    const _Float16* pa1 = A  + (size_t)gr1*K          + kk1;
    const _Float16* pb0 = Wt + (size_t)(nBase+row0)*K + kk0;
    const _Float16* pb1 = Wt + (size_t)(nBase+row1)*K + kk1;
    _Float16* sa0 = &As[row0][kk0];
    _Float16* sa1 = &As[row1][kk1];
    _Float16* sb0 = &Bt[row0][kk0];
    _Float16* sb1 = &Bt[row1][kk1];

    v8f zero = {};
    v8f acc[2][4];
    #pragma unroll
    for (int mi = 0; mi < 2; ++mi)
        #pragma unroll
        for (int ni = 0; ni < 4; ++ni) acc[mi][ni] = zero;

    // prime the pipeline: 4 back-to-back b128 global loads into registers
    uint4 ra0 = *(const uint4*)pa0;
    uint4 ra1 = *(const uint4*)pa1;
    uint4 rb0 = *(const uint4*)pb0;
    uint4 rb1 = *(const uint4*)pb1;

    for (int k0 = 0; k0 < K; k0 += BK) {
        *(uint4*)sa0 = ra0;
        *(uint4*)sa1 = ra1;
        *(uint4*)sb0 = rb0;
        *(uint4*)sb1 = rb1;
        __syncthreads();
        if (k0 + BK < K) {                         // prefetch next tile into regs;
            ra0 = *(const uint4*)(pa0 + k0 + BK);  // overlaps gathers + WMMA below
            ra1 = *(const uint4*)(pa1 + k0 + BK);
            rb0 = *(const uint4*)(pb0 + k0 + BK);
            rb1 = *(const uint4*)(pb1 + k0 + BK);
        }

        union Frag { uint32_t u[8]; v16h v; };
        Frag af[2], bf[4];
        #pragma unroll
        for (int mi = 0; mi < 2; ++mi) {
            int m = wm*32 + mi*16 + laneLo;
            #pragma unroll
            for (int p = 0; p < 8; ++p) {
                int kk = laneHi*16 + p*2;   // contiguous k = 16*laneHi + s (matches B)
                af[mi].u[p] = *(const uint32_t*)&As[m][kk];
            }
        }
        #pragma unroll
        for (int ni = 0; ni < 4; ++ni) {
            int n = wn*64 + ni*16 + laneLo;
            #pragma unroll
            for (int p = 0; p < 8; ++p) {
                bf[ni].u[p] = *(const uint32_t*)&Bt[n][laneHi*16 + p*2];
            }
        }
        #pragma unroll
        for (int mi = 0; mi < 2; ++mi)
            #pragma unroll
            for (int ni = 0; ni < 4; ++ni)
                acc[mi][ni] = __builtin_amdgcn_wmma_f32_16x16x32_f16(
                    false, af[mi].v, false, bf[ni].v,
                    (short)0, acc[mi][ni], false, false);
        __syncthreads();
    }

    // store per C layout: VGPR j -> row (laneHi?8:0)+j, col lane&15
    if (Cf) {
        #pragma unroll
        for (int mi = 0; mi < 2; ++mi)
            #pragma unroll
            for (int ni = 0; ni < 4; ++ni) {
                int n = nBase + wn*64 + ni*16 + laneLo;
                float bb = bias[n];
                #pragma unroll
                for (int j = 0; j < 8; ++j) {
                    int m = mBase + wm*32 + mi*16 + laneHi*8 + j;
                    if (m < M) {
                        float vv = acc[mi][ni][j] + bb;
                        if (relu) vv = fmaxf(vv, 0.0f);
                        Cf[(size_t)m*N + n] = vv;
                    }
                }
            }
    }
    if (Ch) {
        #pragma unroll
        for (int mi = 0; mi < 2; ++mi)
            #pragma unroll
            for (int ni = 0; ni < 4; ++ni) {
                int n = nBase + wn*64 + ni*16 + laneLo;
                float bb = bias[n];
                #pragma unroll
                for (int j = 0; j < 8; ++j) {
                    int m = mBase + wm*32 + mi*16 + laneHi*8 + j;
                    if (m < M) {
                        float vv = acc[mi][ni][j] + bb;
                        if (relu) vv = fmaxf(vv, 0.0f);
                        Ch[(size_t)m*N + n] = (_Float16)vv;
                    }
                }
            }
    }
}

// ---------------- weight convert + transpose: Wt[n][k] = (f16)W[k][n] --------
__global__ __launch_bounds__(256)
void wcvt_kernel(const float* __restrict__ W, _Float16* __restrict__ Wt, int K, int N)
{
    __shared__ float t[32][33];
    const int nb = blockIdx.x * 32, kb = blockIdx.y * 32;
    const int tx = threadIdx.x, ty = threadIdx.y;   // blockDim (32, 8)
    #pragma unroll
    for (int r = 0; r < 4; ++r)
        t[ty + 8*r][tx] = W[(size_t)(kb + ty + 8*r)*N + nb + tx];
    __syncthreads();
    #pragma unroll
    for (int r = 0; r < 4; ++r)
        Wt[(size_t)(nb + ty + 8*r)*K + kb + tx] = (_Float16)t[tx][ty + 8*r];
}

// ---------------- f32 -> f16 convert (layer-0 input) -------------------------
__global__ __launch_bounds__(256)
void cvt_kernel(const float* __restrict__ in, _Float16* __restrict__ out, int n4)
{
    int i = (blockIdx.x * 256 + threadIdx.x);
    if (i < n4) {
        float4 f = *(const float4*)(in + (size_t)i*4);
        union { _Float16 h[4]; uint2 u; } p;
        p.h[0] = (_Float16)f.x; p.h[1] = (_Float16)f.y;
        p.h[2] = (_Float16)f.z; p.h[3] = (_Float16)f.w;
        *(uint2*)(out + (size_t)i*4) = p.u;
    }
}

// ---------------- masked attention: one block per (b, var, head) -------------
__global__ __launch_bounds__(256)
void attn_kernel(const float* __restrict__ Q, const float* __restrict__ Kb,
                 const float* __restrict__ Vb, _Float16* __restrict__ ctx)
{
    __shared__ float Ks[PP][EE+1];
    __shared__ float Vs[PP][EE+1];
    __shared__ float qs[8][EE];
    __shared__ float ps[8][104];

    const int tid  = threadIdx.x;
    const int lane = tid & 31;
    const int wave = tid >> 5;

    int idx = blockIdx.x;
    const int h = idx % HH;    idx /= HH;
    const int v = idx % NVARS; idx /= NVARS;
    const int b = idx;

    const int rowBase = b*LL + v*NTOK;
    const int colBase = h*EE;
    const float scale = 0.125f;   // 1/sqrt(64)

    for (int i = tid; i < PP*EE; i += 256) {
        int k = i >> 6, e = i & 63;
        Ks[k][e] = Kb[(size_t)(rowBase+k)*DDIM + colBase + e];
        Vs[k][e] = Vb[(size_t)(rowBase+k)*DDIM + colBase + e];
    }
    __syncthreads();

    for (int t = 0; t < 13; ++t) {           // 13*8 >= 97 queries, uniform barriers
        int q = t*8 + wave;
        bool active = q < NTOK;
        size_t qrow = (size_t)(rowBase + q)*DDIM + colBase;
        if (active) {
            qs[wave][lane]    = Q[qrow + lane];
            qs[wave][lane+32] = Q[qrow + lane + 32];
        }
        __syncthreads();
        if (active) {
            float s0 = 0.f, s1 = 0.f, s2 = 0.f;
            for (int e = 0; e < EE; ++e) {
                float qe = qs[wave][e];
                s0 += qe * Ks[lane     ][e];
                s1 += qe * Ks[lane + 32][e];
                s2 += qe * Ks[lane + 64][e];
            }
            ps[wave][lane]      = s0*scale;
            ps[wave][lane + 32] = s1*scale;
            ps[wave][lane + 64] = s2*scale;
            if (q == PP && lane < NVARS-1) {   // global token: 6 cross-var globals
                int u = lane + (lane >= v ? 1 : 0);
                size_t krow = (size_t)(b*LL + u*NTOK + PP)*DDIM + colBase;
                float sx = 0.f;
                for (int e = 0; e < EE; ++e) sx += qs[wave][e]*Kb[krow + e];
                ps[wave][PP + lane] = sx*scale;
            }
        }
        __syncthreads();
        if (active) {
            int nsc = (q == PP) ? (PP + NVARS - 1) : PP;
            float mx = -1e30f;
            for (int k = 0; k < nsc; ++k) mx = fmaxf(mx, ps[wave][k]);
            float ssum = 0.f;
            for (int k = 0; k < nsc; ++k) ssum += __expf(ps[wave][k] - mx);
            float inv = 1.0f / ssum;
            #pragma unroll
            for (int e2 = 0; e2 < 2; ++e2) {
                int e = lane + 32*e2;
                float acc = 0.f;
                for (int k = 0; k < PP; ++k)
                    acc += __expf(ps[wave][k] - mx) * Vs[k][e];
                if (q == PP) {
                    for (int j = 0; j < NVARS-1; ++j) {
                        int u = j + (j >= v ? 1 : 0);
                        size_t vrow = (size_t)(b*LL + u*NTOK + PP)*DDIM + colBase;
                        acc += __expf(ps[wave][PP + j] - mx) * Vb[vrow + e];
                    }
                }
                ctx[qrow + e] = (_Float16)(acc * inv);
            }
        }
        __syncthreads();
    }
}

// ---------------- residual-add + LayerNorm, dual f32/f16 output --------------
__global__ __launch_bounds__(256)
void add_ln_kernel(const float* __restrict__ x, const float* __restrict__ y,
                   const float* __restrict__ g, const float* __restrict__ b,
                   float* __restrict__ out, _Float16* __restrict__ outH)
{
    __shared__ float rs[8], rq[8];
    const int tid  = threadIdx.x;
    const int lane = tid & 31;
    const int wave = tid >> 5;
    size_t base = (size_t)blockIdx.x * DDIM;

    float v0 = x[base + tid];
    float v1 = x[base + tid + 256];
    if (y) { v0 += y[base + tid]; v1 += y[base + tid + 256]; }
    float s  = v0 + v1;
    float s2 = v0*v0 + v1*v1;
    #pragma unroll
    for (int m = 16; m > 0; m >>= 1) {
        s  += __shfl_xor(s,  m, 32);
        s2 += __shfl_xor(s2, m, 32);
    }
    if (lane == 0) { rs[wave] = s; rq[wave] = s2; }
    __syncthreads();
    if (tid == 0) {
        float ts = 0.f, tq = 0.f;
        for (int i = 0; i < 8; ++i) { ts += rs[i]; tq += rq[i]; }
        rs[0] = ts; rq[0] = tq;
    }
    __syncthreads();
    float mean = rs[0] * (1.0f/DDIM);
    float var  = rq[0] * (1.0f/DDIM) - mean*mean;
    float inv  = rsqrtf(var + 1e-5f);
    float o0 = (v0 - mean)*inv*g[tid]       + b[tid];
    float o1 = (v1 - mean)*inv*g[tid + 256] + b[tid + 256];
    if (out)  { out[base + tid] = o0;              out[base + tid + 256] = o1; }
    if (outH) { outH[base + tid] = (_Float16)o0;   outH[base + tid + 256] = (_Float16)o1; }
}

// ---------------- host orchestration -----------------------------------------
extern "C" void kernel_launch(void* const* d_in, const int* in_sizes, int n_in,
                              void* d_out, int out_size, void* d_ws, size_t ws_size,
                              hipStream_t stream)
{
    const float* x   = (const float*)d_in[0];
    const float* Wq  = (const float*)d_in[1];  const float* bq  = (const float*)d_in[2];
    const float* Wk  = (const float*)d_in[3];  const float* bk  = (const float*)d_in[4];
    const float* Wv  = (const float*)d_in[5];  const float* bv  = (const float*)d_in[6];
    const float* Wo  = (const float*)d_in[7];  const float* bo  = (const float*)d_in[8];
    const float* W1  = (const float*)d_in[9];  const float* b1  = (const float*)d_in[10];
    const float* W2  = (const float*)d_in[11]; const float* b2  = (const float*)d_in[12];
    const float* g1  = (const float*)d_in[13]; const float* be1 = (const float*)d_in[14];
    const float* g2  = (const float*)d_in[15]; const float* be2 = (const float*)d_in[16];
    const float* gf  = (const float*)d_in[17]; const float* bf  = (const float*)d_in[18];

    const size_t SD = (size_t)MSROW * DDIM;
    char* p = (char*)d_ws;
    auto allocF = [&](size_t n) { float* r = (float*)p;    p += n*sizeof(float);    return r; };
    auto allocH = [&](size_t n) { _Float16* r = (_Float16*)p; p += n*sizeof(_Float16); return r; };

    // f32 buffers
    float* Qb  = allocF(SD);
    float* Kb  = allocF(SD);
    float* Vb  = allocF(SD);
    float* TMP = allocF(SD);
    float* X1  = allocF(SD);
    float* XC  = allocF(SD);
    // f16 activation buffers
    _Float16* Xh   = allocH(SD);
    _Float16* X1h  = allocH(SD);
    _Float16* XCh  = allocH(SD);
    _Float16* CTXh = allocH(SD);
    _Float16* FFh  = allocH((size_t)MSROW * DFF);
    // f16 transposed weights: per layer [Wq Wk Wv Wo | W1 | W2]
    const size_t oQ = 0, oK2 = (size_t)DDIM*DDIM, oV = 2*oK2, oO = 3*oK2;
    const size_t o1 = 4*oK2, o2 = o1 + (size_t)DDIM*DFF;
    const size_t PER = o2 + (size_t)DFF*DDIM;
    _Float16* Wh = allocH(2*PER);

    dim3 blk(256);
    dim3 gD(DDIM/BN, (MSROW + BM - 1)/BM);    // N = 512
    dim3 gF(DFF/BN,  (MSROW + BM - 1)/BM);    // N = 2048
    dim3 tblk(32, 8);

    // one-time per call: convert layer-0 input + all weights (transposed) to f16
    cvt_kernel<<<(int)((SD/4 + 255)/256), blk, 0, stream>>>(x, Xh, (int)(SD/4));
    for (int l = 0; l < 2; ++l) {
        const size_t oDD = (size_t)l*DDIM*DDIM, oDF = (size_t)l*DDIM*DFF;
        _Float16* wl = Wh + (size_t)l*PER;
        wcvt_kernel<<<dim3(DDIM/32, DDIM/32), tblk, 0, stream>>>(Wq+oDD, wl+oQ,  DDIM, DDIM);
        wcvt_kernel<<<dim3(DDIM/32, DDIM/32), tblk, 0, stream>>>(Wk+oDD, wl+oK2, DDIM, DDIM);
        wcvt_kernel<<<dim3(DDIM/32, DDIM/32), tblk, 0, stream>>>(Wv+oDD, wl+oV,  DDIM, DDIM);
        wcvt_kernel<<<dim3(DDIM/32, DDIM/32), tblk, 0, stream>>>(Wo+oDD, wl+oO,  DDIM, DDIM);
        wcvt_kernel<<<dim3(DFF/32,  DDIM/32), tblk, 0, stream>>>(W1+oDF, wl+o1,  DDIM, DFF);
        wcvt_kernel<<<dim3(DDIM/32, DFF/32),  tblk, 0, stream>>>(W2+oDF, wl+o2,  DFF,  DDIM);
    }

    for (int l = 0; l < 2; ++l) {
        const _Float16* xinH = l ? XCh : Xh;
        const float*    xin  = l ? XC  : x;
        const size_t oD = (size_t)l*DDIM, oF = (size_t)l*DFF;
        _Float16* wl = Wh + (size_t)l*PER;

        gemm_bias_kernel<<<gD, blk, 0, stream>>>(xinH, wl+oQ,  bq+oD, Qb, nullptr, MSROW, DDIM, DDIM, 0);
        gemm_bias_kernel<<<gD, blk, 0, stream>>>(xinH, wl+oK2, bk+oD, Kb, nullptr, MSROW, DDIM, DDIM, 0);
        gemm_bias_kernel<<<gD, blk, 0, stream>>>(xinH, wl+oV,  bv+oD, Vb, nullptr, MSROW, DDIM, DDIM, 0);
        attn_kernel<<<BB*NVARS*HH, blk, 0, stream>>>(Qb, Kb, Vb, CTXh);
        gemm_bias_kernel<<<gD, blk, 0, stream>>>(CTXh, wl+oO, bo+oD, TMP, nullptr, MSROW, DDIM, DDIM, 0);
        add_ln_kernel<<<MSROW, blk, 0, stream>>>(xin, TMP, g1+oD, be1+oD, X1, X1h);
        gemm_bias_kernel<<<gF, blk, 0, stream>>>(X1h, wl+o1, b1+oF, nullptr, FFh, MSROW, DFF, DDIM, 1);
        gemm_bias_kernel<<<gD, blk, 0, stream>>>(FFh, wl+o2, b2+oD, TMP, nullptr, MSROW, DDIM, DFF, 0);
        add_ln_kernel<<<MSROW, blk, 0, stream>>>(X1, TMP, g2+oD, be2+oD, XC, XCh);
    }
    add_ln_kernel<<<MSROW, blk, 0, stream>>>(XC, nullptr, gf, bf, (float*)d_out, nullptr);
}
